// InfoVAE_13228499272255
// MI455X (gfx1250) — matrix operand order repaired and checked by reading, not compiled, not run on hardware
//
#include <hip/hip_runtime.h>
#include <math.h>

typedef unsigned int   u32;
typedef unsigned short u16;
typedef __attribute__((ext_vector_type(16))) __bf16 bf16x16;
typedef __attribute__((ext_vector_type(8)))  float  f32x8;
typedef __attribute__((ext_vector_type(4)))  u32    u32x4;
typedef __attribute__((ext_vector_type(2)))  u32    u32x2;

#define DEV __device__ __forceinline__

DEV u16 f2bf(float f) {                       // f32 -> bf16, round to nearest even
  u32 u = __builtin_bit_cast(u32, f);
  u += 0x7FFFu + ((u >> 16) & 1u);
  return (u16)(u >> 16);
}
DEV float bf2f(u16 h) {
  u32 u = ((u32)h) << 16;
  return __builtin_bit_cast(float, u);
}

// ---- CDNA5 async global->LDS copy (ASYNCcnt path, ISA 08_async_tensor) ----
// LDS destination address = low 32 bits of the generic shared pointer
// (generic LDS addr = {aperture, offset[31:0]}, ISA 00_overview 10.2).
DEV void async_load_b128(const void* lds_dst, const void* gsrc) {
  u32 laddr = (u32)(size_t)lds_dst;
  asm volatile("global_load_async_to_lds_b128 %0, %1, off"
               :: "v"(laddr), "v"(gsrc) : "memory");
}
DEV void wait_async0() {
  asm volatile("s_wait_asynccnt 0" ::: "memory");
}

union FragBF { u32x4 q[2]; bf16x16 v; };

// A-fragment (16x32 bf16, ISA 7.12.2): lanes 0-15 (rows M=0..15) hold K chunks
// {0..7, 16..23}; lanes 16-31 hold {8..15, 24..31}. Two ds_load_b128 per lane.
DEV void load_frag_a(FragBF& fr, const u16* row_ptr, int lane, int kbase) {
  int b = kbase + ((lane < 16) ? 0 : 8);
  fr.q[0] = *reinterpret_cast<const u32x4*>(row_ptr + b);
  fr.q[1] = *reinterpret_cast<const u32x4*>(row_ptr + b + 16);
}
// B-fragment (32x16 bf16), stored K-major per column in LDS: lanes 0-15
// (col N) hold K chunks {0..7, 16..23}; lanes 16-31 hold {8..15, 24..31}.
DEV void load_frag_b(FragBF& fr, const u16* row_ptr, int lane, int kbase) {
  int b = kbase + ((lane < 16) ? 0 : 8);
  fr.q[0] = *reinterpret_cast<const u32x4*>(row_ptr + b);
  fr.q[1] = *reinterpret_cast<const u32x4*>(row_ptr + b + 16);
}

DEV f32x8 wmma_bf16(const FragBF& a, const FragBF& b, f32x8 c) {
  return __builtin_amdgcn_wmma_f32_16x16x32_bf16(false, a.v, false, b.v,
                                                 (short)0, c, false, false);
}

enum { ACT_LRELU = 0, ACT_TANH = 1, ACT_LRELU_CLIP = 2 };
enum { OUT_BF16 = 0, OUT_F32 = 1, OUT_LOSS = 2 };

DEV float apply_act(float v, int act) {
  if (act == ACT_TANH) return tanhf(v);
  float r = (v > 0.0f) ? v : 0.01f * v;
  if (act == ACT_LRELU_CLIP) r = fminf(fmaxf(r, -10.0f), 10.0f);
  return r;
}

// One-time f32 -> bf16 conversion (inputs are L2-resident afterwards).
__global__ void cvt_f32_bf16(const float* __restrict__ src,
                             u16* __restrict__ dst, int n4)
{
  int i = blockIdx.x * blockDim.x + threadIdx.x;
  if (i < n4) {
    float4 f = reinterpret_cast<const float4*>(src)[i];
    u32x2 p;
    p.x = (u32)f2bf(f.x) | ((u32)f2bf(f.y) << 16);
    p.y = (u32)f2bf(f.z) | ((u32)f2bf(f.w) << 16);
    reinterpret_cast<u32x2*>(dst)[i] = p;
  }
}

// C = act(A[M,K] @ B[K,N] + bias). A, B are bf16; output is bf16, f32, or a
// fused sum((C - Xref)^2) per-block partial. 256 threads = 8 wave32 (2x4 wave
// grid), each wave owns 64x(16*WNT). Double-buffered LDS, BK=32; A-tile is
// copied with async global->LDS (no VGPR staging), B-tile is repacked K-major.
template<int BN, int ACT, int OUT>
__global__ __launch_bounds__(256)
void gemm_bias_act(const u16* __restrict__ A, const u16* __restrict__ Bm,
                   const float* __restrict__ bias,
                   u16* __restrict__ Cb, float* __restrict__ Cf,
                   const float* __restrict__ Xref, float* __restrict__ partials,
                   int M, int N, int K)
{
  constexpr int BM  = 128, BK = 32;
  constexpr int AS  = BK + 8;                 // LDS row stride (u16): 80B, 16B-aligned
  constexpr int WNT = BN / 64;                // 16-col tiles per wave (2 or 1)
  constexpr int ACH = (BM * BK) / 8 / 256;    // 16B async chunks per thread (2)
  constexpr int NB  = (BK * BN) / 8 / 256;    // 16B B chunks per thread (2 or 1)
  constexpr int BC8 = BN / 8;                 // 8-elem chunks per B row

  __shared__ alignas(16) u16 sA[2][BM * AS];
  __shared__ alignas(16) u16 sB[2][BN * AS];

  const int tid  = threadIdx.x;
  const int lane = tid & 31;
  const int wid  = tid >> 5;
  const int wm   = wid >> 2;                  // 0..1 : 64 rows each
  const int wn   = wid & 3;                   // 0..3
  const int m0   = blockIdx.y * BM;
  const int n0   = blockIdx.x * BN;
  const int nb   = wn * 16 * WNT;
  (void)M;

  f32x8 acc[4][WNT];
  #pragma unroll
  for (int mt = 0; mt < 4; ++mt)
    #pragma unroll
    for (int nt = 0; nt < WNT; ++nt)
      #pragma unroll
      for (int r = 0; r < 8; ++r) acc[mt][nt][r] = 0.0f;

  u32x4 rb[NB];
  const int KT = K / BK;

  auto stage_a_async = [&](int kt, int buf) {
    const int k0 = kt * BK;
    #pragma unroll
    for (int i = 0; i < ACH; ++i) {
      int idx = tid + i * 256;
      int row = idx >> 2;                     // 4 x 16B chunks per 32-elem row
      int off = (idx & 3) * 8;                // u16 units
      async_load_b128(&sA[buf][row * AS + off],
                      A + (size_t)(m0 + row) * K + (k0 + off));
    }
  };
  auto load_b = [&](int kt) {
    const int k0 = kt * BK;
    #pragma unroll
    for (int i = 0; i < NB; ++i) {
      int idx = tid + i * 256;
      int row = idx / BC8;                    // k
      int col = (idx % BC8) * 8;              // n
      rb[i] = *reinterpret_cast<const u32x4*>(Bm + (size_t)(k0 + row) * N + (n0 + col));
    }
  };
  auto store_b = [&](int buf) {               // repack K-major (transpose)
    #pragma unroll
    for (int i = 0; i < NB; ++i) {
      int idx = tid + i * 256;
      int row = idx / BC8;
      int col = (idx % BC8) * 8;
      u32 w0 = rb[i].x, w1 = rb[i].y, w2 = rb[i].z, w3 = rb[i].w;
      sB[buf][(col + 0) * AS + row] = (u16)(w0 & 0xffff);
      sB[buf][(col + 1) * AS + row] = (u16)(w0 >> 16);
      sB[buf][(col + 2) * AS + row] = (u16)(w1 & 0xffff);
      sB[buf][(col + 3) * AS + row] = (u16)(w1 >> 16);
      sB[buf][(col + 4) * AS + row] = (u16)(w2 & 0xffff);
      sB[buf][(col + 5) * AS + row] = (u16)(w2 >> 16);
      sB[buf][(col + 6) * AS + row] = (u16)(w3 & 0xffff);
      sB[buf][(col + 7) * AS + row] = (u16)(w3 >> 16);
    }
  };

  stage_a_async(0, 0);
  load_b(0);
  store_b(0);
  wait_async0();
  __syncthreads();

  for (int kt = 0; kt < KT; ++kt) {
    if (kt + 1 < KT) {                        // async fill of the other buffer
      stage_a_async(kt + 1, (kt + 1) & 1);
      load_b(kt + 1);
    }

    const u16* pa = sA[kt & 1];
    const u16* pb = sB[kt & 1];
    FragBF bfr[WNT];
    #pragma unroll
    for (int nt = 0; nt < WNT; ++nt)
      load_frag_b(bfr[nt], pb + (size_t)(nb + nt * 16 + (lane & 15)) * AS, lane, 0);
    #pragma unroll
    for (int mt = 0; mt < 4; ++mt) {
      FragBF afr;
      load_frag_a(afr, pa + (size_t)(wm * 64 + mt * 16 + (lane & 15)) * AS, lane, 0);
      #pragma unroll
      for (int nt = 0; nt < WNT; ++nt)
        acc[mt][nt] = wmma_bf16(afr, bfr[nt], acc[mt][nt]);
    }

    if (kt + 1 < KT) store_b((kt + 1) & 1);
    wait_async0();
    __syncthreads();
  }

  // Epilogue: bias + activation, then bf16 store, f32 store, or fused loss.
  float lsum = 0.0f;
  #pragma unroll
  for (int mt = 0; mt < 4; ++mt) {
    #pragma unroll
    for (int nt = 0; nt < WNT; ++nt) {
      const int n  = n0 + nb + nt * 16 + (lane & 15);
      const float bv = bias[n];
      #pragma unroll
      for (int r = 0; r < 8; ++r) {
        const int m = m0 + wm * 64 + mt * 16 + (lane >> 4) * 8 + r;
        float v = apply_act(acc[mt][nt][r] + bv, ACT);
        if constexpr (OUT == OUT_BF16) {
          Cb[(size_t)m * N + n] = f2bf(v);
        } else if constexpr (OUT == OUT_F32) {
          Cf[(size_t)m * N + n] = v;
        } else {
          float d = v - Xref[(size_t)m * N + n];
          lsum += d * d;
        }
      }
    }
  }
  if constexpr (OUT == OUT_LOSS) {
    __shared__ float red[256];
    red[tid] = lsum;
    __syncthreads();
    for (int s = 128; s > 0; s >>= 1) {
      if (tid < s) red[tid] += red[tid + s];
      __syncthreads();
    }
    if (tid == 0) partials[blockIdx.y * gridDim.x + blockIdx.x] = red[0];
  }
}

// One 128x128 tile of the RBF Gram sum: G = X_tile @ Y_tile^T via WMMA
// (K=64 = 2 k-steps). Both tiles are bf16 row-major -> pure async LDS copies.
__global__ __launch_bounds__(256)
void mmd_partial(const u16* __restrict__ X, const u16* __restrict__ Y,
                 float* __restrict__ partials)
{
  constexpr int BT = 128, D = 64, AS = D + 8;   // 144B rows, 16B-aligned
  __shared__ alignas(16) u16 sX[BT * AS];
  __shared__ alignas(16) u16 sY[BT * AS];
  __shared__ float xn[BT], yn[BT];
  __shared__ float red[256];

  const int tid  = threadIdx.x;
  const int lane = tid & 31;
  const int wid  = tid >> 5;
  const int wm   = wid >> 2;
  const int wn   = wid & 3;
  const int i0   = blockIdx.y * BT;
  const int j0   = blockIdx.x * BT;

  #pragma unroll
  for (int i = 0; i < 4; ++i) {               // 1024 16B chunks per tile
    int idx = tid + i * 256;
    int row = idx >> 3;                       // 8 chunks per 64-elem row
    int off = (idx & 7) * 8;                  // u16 units
    async_load_b128(&sX[row * AS + off], X + (size_t)(i0 + row) * D + off);
    async_load_b128(&sY[row * AS + off], Y + (size_t)(j0 + row) * D + off);
  }
  wait_async0();
  __syncthreads();

  if (tid < BT) {                             // row norms from the staged tiles
    float s = 0.0f;
    for (int k = 0; k < D; ++k) { float v = bf2f(sX[tid * AS + k]); s += v * v; }
    xn[tid] = s;
  } else {
    int t = tid - BT;
    float s = 0.0f;
    for (int k = 0; k < D; ++k) { float v = bf2f(sY[t * AS + k]); s += v * v; }
    yn[t] = s;
  }
  __syncthreads();

  f32x8 acc[4][2];
  #pragma unroll
  for (int mt = 0; mt < 4; ++mt)
    #pragma unroll
    for (int nt = 0; nt < 2; ++nt)
      #pragma unroll
      for (int r = 0; r < 8; ++r) acc[mt][nt][r] = 0.0f;

  #pragma unroll
  for (int ks = 0; ks < 2; ++ks) {
    FragBF bfr[2];
    #pragma unroll
    for (int nt = 0; nt < 2; ++nt)
      load_frag_b(bfr[nt], sY + (size_t)(wn * 32 + nt * 16 + (lane & 15)) * AS, lane, ks * 32);
    #pragma unroll
    for (int mt = 0; mt < 4; ++mt) {
      FragBF afr;
      load_frag_a(afr, sX + (size_t)(wm * 64 + mt * 16 + (lane & 15)) * AS, lane, ks * 32);
      #pragma unroll
      for (int nt = 0; nt < 2; ++nt)
        acc[mt][nt] = wmma_bf16(afr, bfr[nt], acc[mt][nt]);
    }
  }

  const float cexp = -1.0f / 4096.0f;         // 1/(d*d), d = 64
  float lsum = 0.0f;
  #pragma unroll
  for (int mt = 0; mt < 4; ++mt) {
    #pragma unroll
    for (int nt = 0; nt < 2; ++nt) {
      const int j = wn * 32 + nt * 16 + (lane & 15);
      const float yv = yn[j];
      #pragma unroll
      for (int r = 0; r < 8; ++r) {
        const int i = wm * 64 + mt * 16 + (lane >> 4) * 8 + r;
        float s = xn[i] + yv - 2.0f * acc[mt][nt][r];
        lsum += __expf(s * cexp);
      }
    }
  }
  red[tid] = lsum;
  __syncthreads();
  for (int s = 128; s > 0; s >>= 1) {
    if (tid < s) red[tid] += red[tid + s];
    __syncthreads();
  }
  if (tid == 0) partials[blockIdx.y * gridDim.x + blockIdx.x] = red[0];
}

__global__ void make_z_kernel(const float* __restrict__ mean_,
                              const float* __restrict__ lv,
                              const float* __restrict__ ne,
                              const float* __restrict__ nz,
                              u16* __restrict__ z_enc_b,
                              u16* __restrict__ z_b, int n)
{
  int i = blockIdx.x * blockDim.x + threadIdx.x;
  if (i < n) {
    float s = __expf(0.5f * lv[i]);
    float m = mean_[i];
    z_enc_b[i] = f2bf(m + s * ne[i]);
    z_b[i]     = f2bf(m + s * nz[i]);
  }
}

__global__ __launch_bounds__(256)
void finalize_kernel(const float* __restrict__ rp, int nr,
                     const float* __restrict__ kxx, const float* __restrict__ kyy,
                     const float* __restrict__ kxy, int nk,
                     float* __restrict__ out, float inv_recon, float inv_k)
{
  __shared__ float red[256];
  const int tid = threadIdx.x;
  auto reduce = [&](const float* p, int n) -> float {
    float s = 0.0f;
    for (int i = tid; i < n; i += 256) s += p[i];
    red[tid] = s;
    __syncthreads();
    for (int st = 128; st > 0; st >>= 1) {
      if (tid < st) red[tid] += red[tid + st];
      __syncthreads();
    }
    float r = red[0];
    __syncthreads();
    return r;
  };
  float recon = reduce(rp, nr);
  float sxx   = reduce(kxx, nk);
  float syy   = reduce(kyy, nk);
  float sxy   = reduce(kxy, nk);
  if (tid == 0)
    out[0] = recon * inv_recon + 10.0f * (sxx + syy - 2.0f * sxy) * inv_k;
}

extern "C" void kernel_launch(void* const* d_in, const int* in_sizes, int n_in,
                              void* d_out, int out_size, void* d_ws, size_t ws_size,
                              hipStream_t stream)
{
  (void)in_sizes; (void)n_in; (void)out_size; (void)ws_size;
  const int B = 8192, H0 = 1024, H1 = 512, H2 = 64;
  const float* x         = (const float*)d_in[0];
  const float* noise_enc = (const float*)d_in[1];
  const float* noise_z   = (const float*)d_in[2];
  const float* z_prior   = (const float*)d_in[3];
  const float* w1_mu  = (const float*)d_in[4];
  const float* b1_mu  = (const float*)d_in[5];
  const float* w2_mu  = (const float*)d_in[6];
  const float* b2_mu  = (const float*)d_in[7];
  const float* w1_sig = (const float*)d_in[8];
  const float* b1_sig = (const float*)d_in[9];
  const float* w2_sig = (const float*)d_in[10];
  const float* b2_sig = (const float*)d_in[11];
  const float* w1_dec = (const float*)d_in[12];
  const float* b1_dec = (const float*)d_in[13];
  const float* w2_dec = (const float*)d_in[14];
  const float* b2_dec = (const float*)d_in[15];

  char* wsb = (char*)d_ws;
  size_t o = 0;
  auto alloc = [&](size_t bytes) -> char* {
    char* p = wsb + o;
    o += (bytes + 255) & ~(size_t)255;
    return p;
  };
  u16* xb      = (u16*)alloc((size_t)B * H0 * 2);
  u16* w1mu_b  = (u16*)alloc((size_t)H0 * H1 * 2);
  u16* w1sig_b = (u16*)alloc((size_t)H0 * H1 * 2);
  u16* w2mu_b  = (u16*)alloc((size_t)H1 * H2 * 2);
  u16* w2sig_b = (u16*)alloc((size_t)H1 * H2 * 2);
  u16* w1dec_b = (u16*)alloc((size_t)H2 * H1 * 2);
  u16* w2dec_b = (u16*)alloc((size_t)H1 * H0 * 2);
  u16* zp_b    = (u16*)alloc((size_t)B * H2 * 2);
  u16* h1mu_b  = (u16*)alloc((size_t)B * H1 * 2);
  u16* h1sig_b = (u16*)alloc((size_t)B * H1 * 2);
  float* mean_ = (float*)alloc((size_t)B * H2 * 4);
  float* lv    = (float*)alloc((size_t)B * H2 * 4);
  u16* zenc_b  = (u16*)alloc((size_t)B * H2 * 2);
  u16* z_b     = (u16*)alloc((size_t)B * H2 * 2);
  float* recon_p = (float*)alloc(512 * 4);
  float* kxx_p = (float*)alloc(4096 * 4);
  float* kyy_p = (float*)alloc(4096 * 4);
  float* kxy_p = (float*)alloc(4096 * 4);
  u16* h1dec_b = h1mu_b;   // h1_mu fully consumed before dec1 runs (stream order)

  dim3 blk(256);
  auto cvt = [&](const float* s, u16* d, int n) {
    cvt_f32_bf16<<<dim3((n / 4 + 255) / 256), blk, 0, stream>>>(s, d, n / 4);
  };
  // One-time bf16 residency for everything the WMMA pipeline touches.
  cvt(x, xb, B * H0);
  cvt(w1_mu, w1mu_b, H0 * H1);
  cvt(w1_sig, w1sig_b, H0 * H1);
  cvt(w2_mu, w2mu_b, H1 * H2);
  cvt(w2_sig, w2sig_b, H1 * H2);
  cvt(w1_dec, w1dec_b, H2 * H1);
  cvt(w2_dec, w2dec_b, H1 * H0);
  cvt(z_prior, zp_b, B * H2);

  // Encoder layer 1: 8192x1024 @ 1024x512 (bf16 in, bf16 out)
  gemm_bias_act<128, ACT_LRELU, OUT_BF16><<<dim3(H1 / 128, B / 128), blk, 0, stream>>>(
      xb, w1mu_b, b1_mu, h1mu_b, nullptr, nullptr, nullptr, B, H1, H0);
  gemm_bias_act<128, ACT_TANH, OUT_BF16><<<dim3(H1 / 128, B / 128), blk, 0, stream>>>(
      xb, w1sig_b, b1_sig, h1sig_b, nullptr, nullptr, nullptr, B, H1, H0);
  // Encoder layer 2: 8192x512 @ 512x64 (f32 out for reparameterization)
  gemm_bias_act<64, ACT_LRELU, OUT_F32><<<dim3(H2 / 64, B / 128), blk, 0, stream>>>(
      h1mu_b, w2mu_b, b2_mu, nullptr, mean_, nullptr, nullptr, B, H2, H1);
  gemm_bias_act<64, ACT_LRELU_CLIP, OUT_F32><<<dim3(H2 / 64, B / 128), blk, 0, stream>>>(
      h1sig_b, w2sig_b, b2_sig, nullptr, lv, nullptr, nullptr, B, H2, H1);
  // Reparameterize (writes bf16 z_enc, z)
  make_z_kernel<<<dim3((B * H2) / 256), blk, 0, stream>>>(
      mean_, lv, noise_enc, noise_z, zenc_b, z_b, B * H2);
  // Decoder layer 1: 8192x64 @ 64x512
  gemm_bias_act<128, ACT_LRELU, OUT_BF16><<<dim3(H1 / 128, B / 128), blk, 0, stream>>>(
      zenc_b, w1dec_b, b1_dec, h1dec_b, nullptr, nullptr, nullptr, B, H1, H2);
  // Decoder layer 2 fused with recon_loss partials: 8192x512 @ 512x1024
  gemm_bias_act<128, ACT_LRELU, OUT_LOSS><<<dim3(H0 / 128, B / 128), blk, 0, stream>>>(
      h1dec_b, w2dec_b, b2_dec, nullptr, nullptr, x, recon_p, B, H0, H1);
  // MMD Gram sums
  mmd_partial<<<dim3(64, 64), blk, 0, stream>>>(z_b, z_b, kxx_p);
  mmd_partial<<<dim3(64, 64), blk, 0, stream>>>(zp_b, zp_b, kyy_p);
  mmd_partial<<<dim3(64, 64), blk, 0, stream>>>(z_b, zp_b, kxy_p);
  // Deterministic final reduction
  finalize_kernel<<<1, blk, 0, stream>>>(
      recon_p, 512, kxx_p, kyy_p, kxy_p, 4096, (float*)d_out,
      1.0f / ((float)B * (float)H0), 1.0f / ((float)B * (float)B));
}
